// GPS_ZINC_v2_77008763617632
// MI455X (gfx1250) — compile-verified, matrix operand order checked by reading, not run on hardware
//
#include <hip/hip_runtime.h>

// ---------------------------------------------------------------------------
// GraphGPS (ZINC) forward for MI455X / gfx1250 — wave32 + WMMA bf16 GEMMs.
// All dense GEMMs run through v_wmma_f32_16x16x32_bf16 (fp32 accumulate).
// Fragment loads are 16-byte (b128) vectorized; GEMM waves compute 2x2 tiles.
// ---------------------------------------------------------------------------

#define BATCH  4096
#define LMAX   32
#define DMODEL 128
#define NHEAD  4
#define DHEAD  32
#define NLAYER 4
#define FFDIM  256

typedef __attribute__((ext_vector_type(16))) __bf16 v16bf;
typedef __attribute__((ext_vector_type(8)))  float  v8f;

// 16 bf16 fragment = two 16-byte runs (ISA 7.12.2 16-bit A/B layout:
// element j -> K = kb + (j<8 ? j : j+8) + 8*(lane>=16), i.e. runs
// [kb+half*8, +8) and [kb+16+half*8, +8)).
union Frag {
  uint4 u[2];
  v16bf v;
};

// ---- bf16 helpers (round-to-nearest-even) ---------------------------------
__device__ __forceinline__ unsigned short f2bf_bits(float f) {
  union { float f; unsigned int u; } c; c.f = f;
  unsigned int u = c.u + 0x7FFFu + ((c.u >> 16) & 1u);
  return (unsigned short)(u >> 16);
}

// ---------------------------------------------------------------------------
// WMMA GEMM:  C[M,Nc] = act( A_bf16[M,K] @ W_bf16[Nc,K]^T + bias[Nc] )
// One wave computes a 32x32 output block (2x2 WMMA tiles): per K-chunk it
// loads 2 A-frags + 2 B-frags (4x b128-pairs) and issues 4 WMMAs.
// Requires K % 32 == 0 and (Nc/16) even (true for all call sites).
// ---------------------------------------------------------------------------
__global__ __launch_bounds__(128) void wmma_gemm_bias_kernel(
    const unsigned short* __restrict__ A, const unsigned short* __restrict__ W,
    const float* __restrict__ bias, float* __restrict__ C,
    int M, int Nc, int K, int relu)
{
  const int lane = threadIdx.x & 31;
  const int wave = threadIdx.x >> 5;
  const int tilesM2 = (M + 31) >> 5;       // 32-row blocks
  const int tilesN2 = Nc >> 5;             // 32-col blocks
  const int blk = blockIdx.x * 4 + wave;
  if (blk >= tilesM2 * tilesN2) return;
  const int bm = blk / tilesN2;
  const int bn = blk - bm * tilesN2;

  const int half  = lane >> 4;
  const int lidx  = lane & 15;
  const int half8 = half * 8;

  const int rowA0 = bm * 32 + lidx;
  const int rowA1 = rowA0 + 16;
  // clamp OOB rows to row 0: their results are never stored
  const unsigned short* Arow0 = A + (size_t)(rowA0 < M ? rowA0 : 0) * K;
  const unsigned short* Arow1 = A + (size_t)(rowA1 < M ? rowA1 : 0) * K;
  const unsigned short* Wrow0 = W + (size_t)(bn * 32 + lidx) * K;
  const unsigned short* Wrow1 = W + (size_t)(bn * 32 + 16 + lidx) * K;

  v8f acc00 = {}, acc01 = {}, acc10 = {}, acc11 = {};
  for (int kb = 0; kb < K; kb += 32) {
    Frag a0, a1, b0, b1;
    a0.u[0] = *(const uint4*)(Arow0 + kb + half8);
    a0.u[1] = *(const uint4*)(Arow0 + kb + 16 + half8);
    a1.u[0] = *(const uint4*)(Arow1 + kb + half8);
    a1.u[1] = *(const uint4*)(Arow1 + kb + 16 + half8);
    b0.u[0] = *(const uint4*)(Wrow0 + kb + half8);
    b0.u[1] = *(const uint4*)(Wrow0 + kb + 16 + half8);
    b1.u[0] = *(const uint4*)(Wrow1 + kb + half8);
    b1.u[1] = *(const uint4*)(Wrow1 + kb + 16 + half8);
    acc00 = __builtin_amdgcn_wmma_f32_16x16x32_bf16(false, a0.v, false, b0.v, (short)0, acc00, false, false);
    acc01 = __builtin_amdgcn_wmma_f32_16x16x32_bf16(false, a0.v, false, b1.v, (short)0, acc01, false, false);
    acc10 = __builtin_amdgcn_wmma_f32_16x16x32_bf16(false, a1.v, false, b0.v, (short)0, acc10, false, false);
    acc11 = __builtin_amdgcn_wmma_f32_16x16x32_bf16(false, a1.v, false, b1.v, (short)0, acc11, false, false);
  }

  // C/D layout: lane -> col = lidx (+tile), vgpr r -> row r + 8*half (+tile)
  const int colBase = bn * 32;
  const float bias0 = bias[colBase + lidx];
  const float bias1 = bias[colBase + 16 + lidx];
#pragma unroll
  for (int r = 0; r < 8; ++r) {
    const int row0 = bm * 32 + r + half * 8;
    const int row1 = row0 + 16;
    if (row0 < M) {
      float v0 = acc00[r] + bias0;
      float v1 = acc01[r] + bias1;
      if (relu) { v0 = fmaxf(v0, 0.0f); v1 = fmaxf(v1, 0.0f); }
      C[(size_t)row0 * Nc + colBase + lidx] = v0;
      C[(size_t)row0 * Nc + colBase + 16 + lidx] = v1;
    }
    if (row1 < M) {
      float v0 = acc10[r] + bias0;
      float v1 = acc11[r] + bias1;
      if (relu) { v0 = fmaxf(v0, 0.0f); v1 = fmaxf(v1, 0.0f); }
      C[(size_t)row1 * Nc + colBase + lidx] = v0;
      C[(size_t)row1 * Nc + colBase + 16 + lidx] = v1;
    }
  }
}

// ---------------------------------------------------------------------------
// Block-diagonal masked multi-head attention, one graph per block.
// 4 waves = 4 heads; scores and P@V both via WMMA bf16 (K = DH = L = 32).
// V is staged transposed so every fragment load is two contiguous b128s.
// ---------------------------------------------------------------------------
__global__ __launch_bounds__(128) void attention_kernel(
    const float* __restrict__ qkv, const int* __restrict__ inv,
    float* __restrict__ o)
{
  __shared__ unsigned short q_s[NHEAD][LMAX][DHEAD];  // later reused for probs
  __shared__ unsigned short k_s[NHEAD][LMAX][DHEAD];
  __shared__ unsigned short v_t[NHEAD][DHEAD][LMAX];  // transposed: [d][key]
  __shared__ float sc[NHEAD][LMAX][LMAX];
  __shared__ float kmask[LMAX];
  __shared__ int   nodes[LMAX];

  const int b   = blockIdx.x;
  const int tid = threadIdx.x;

  if (tid < LMAX) {
    const int nd = inv[b * LMAX + tid];
    nodes[tid] = nd;
    kmask[tid] = (nd >= 0) ? 0.0f : -1e9f;
  }
  __syncthreads();

  const float scale = 0.17677669529663687f;  // 1/sqrt(DH)
  for (int idx = tid; idx < LMAX * DMODEL; idx += 128) {
    const int s = idx >> 7;
    const int c = idx & 127;
    const int h = c >> 5;
    const int d = c & 31;
    const int nd = nodes[s];
    float qv = 0.f, kv = 0.f, vv = 0.f;
    if (nd >= 0) {
      const float* base = qkv + (size_t)nd * (3 * DMODEL);
      qv = base[c] * scale;
      kv = base[DMODEL + c];
      vv = base[2 * DMODEL + c];
    }
    q_s[h][s][d] = f2bf_bits(qv);
    k_s[h][s][d] = f2bf_bits(kv);
    v_t[h][d][s] = f2bf_bits(vv);
  }
  __syncthreads();

  const int h     = tid >> 5;
  const int lane  = tid & 31;
  const int half  = lane >> 4;
  const int lidx  = lane & 15;
  const int half8 = half * 8;

  // scores = Q @ K^T  (2x2 tiles of 16x16, one WMMA each, K=32)
  for (int t = 0; t < 4; ++t) {
    const int tm = t >> 1, tn = t & 1;
    Frag a, bf;
    a.u[0]  = *(const uint4*)&q_s[h][tm * 16 + lidx][half8];
    a.u[1]  = *(const uint4*)&q_s[h][tm * 16 + lidx][16 + half8];
    bf.u[0] = *(const uint4*)&k_s[h][tn * 16 + lidx][half8];
    bf.u[1] = *(const uint4*)&k_s[h][tn * 16 + lidx][16 + half8];
    v8f c = {};
    c = __builtin_amdgcn_wmma_f32_16x16x32_bf16(false, a.v, false, bf.v,
                                                (short)0, c, false, false);
#pragma unroll
    for (int r = 0; r < 8; ++r)
      sc[h][tm * 16 + r + half * 8][tn * 16 + lidx] = c[r];
  }
  __syncthreads();

  // softmax over keys (128 threads = 4 heads x 32 rows); probs -> q_s (bf16)
  {
    const int hh  = tid >> 5;
    const int row = tid & 31;
    float vals[LMAX];
    float mx = -1e30f;
#pragma unroll
    for (int c = 0; c < LMAX; ++c) {
      const float v = sc[hh][row][c] + kmask[c];
      vals[c] = v; mx = fmaxf(mx, v);
    }
    float sum = 0.f;
#pragma unroll
    for (int c = 0; c < LMAX; ++c) { const float e = __expf(vals[c] - mx); vals[c] = e; sum += e; }
    const float is = 1.0f / sum;
#pragma unroll
    for (int c = 0; c < LMAX; ++c) q_s[hh][row][c] = f2bf_bits(vals[c] * is);
  }
  __syncthreads();

  // O = P @ V  (K = 32 keys); B[k][n] = V[k][n] = v_t[h][n][k] (contiguous in k)
  for (int t = 0; t < 4; ++t) {
    const int tm = t >> 1, tn = t & 1;
    Frag a, bf;
    a.u[0]  = *(const uint4*)&q_s[h][tm * 16 + lidx][half8];
    a.u[1]  = *(const uint4*)&q_s[h][tm * 16 + lidx][16 + half8];
    bf.u[0] = *(const uint4*)&v_t[h][tn * 16 + lidx][half8];
    bf.u[1] = *(const uint4*)&v_t[h][tn * 16 + lidx][16 + half8];
    v8f c = {};
    c = __builtin_amdgcn_wmma_f32_16x16x32_bf16(false, a.v, false, bf.v,
                                                (short)0, c, false, false);
#pragma unroll
    for (int r = 0; r < 8; ++r) {
      const int row = tm * 16 + r + half * 8;
      const int nd  = nodes[row];
      if (nd >= 0)
        o[(size_t)nd * DMODEL + h * DHEAD + tn * 16 + lidx] = c[r];
    }
  }
}

// ---------------------------------------------------------------------------
// Small utility kernels
// ---------------------------------------------------------------------------
__global__ void zero_f32_kernel(float* p, size_t n) {
  size_t i = (size_t)blockIdx.x * blockDim.x + threadIdx.x;
  if (i < n) p[i] = 0.0f;
}
__global__ void fill_i32_kernel(int* p, int v, size_t n) {
  size_t i = (size_t)blockIdx.x * blockDim.x + threadIdx.x;
  if (i < n) p[i] = v;
}
__global__ void f32_to_bf16_kernel(const float* __restrict__ s, unsigned short* __restrict__ d, size_t n) {
  size_t i = (size_t)blockIdx.x * blockDim.x + threadIdx.x;
  if (i < n) d[i] = f2bf_bits(s[i]);
}
__global__ void add_to_bf16_kernel(const float* __restrict__ a, const float* __restrict__ b,
                                   unsigned short* __restrict__ d, size_t n) {
  size_t i = (size_t)blockIdx.x * blockDim.x + threadIdx.x;
  if (i < n) d[i] = f2bf_bits(a[i] + b[i]);
}
__global__ void residual3_kernel(float* __restrict__ x, const float* __restrict__ g,
                                 const float* __restrict__ o, size_t n) {
  size_t i = (size_t)blockIdx.x * blockDim.x + threadIdx.x;
  if (i < n) x[i] = x[i] + fmaxf(g[i], 0.0f) + o[i];
}
__global__ void add_inplace_kernel(float* __restrict__ x, const float* __restrict__ f, size_t n) {
  size_t i = (size_t)blockIdx.x * blockDim.x + threadIdx.x;
  if (i < n) x[i] += f[i];
}
__global__ void pad_inw_kernel(const float* __restrict__ w, unsigned short* __restrict__ d) {
  int i = blockIdx.x * blockDim.x + threadIdx.x;
  if (i >= DMODEL * 96) return;
  const int o = i / 96, c = i % 96;
  d[i] = (c < 80) ? f2bf_bits(w[o * 80 + c]) : (unsigned short)0;
}
__global__ void build_input_kernel(const int* __restrict__ atom_idx, const float* __restrict__ atom_emb,
                                   const float* __restrict__ peh, unsigned short* __restrict__ d, int Nn) {
  size_t i = (size_t)blockIdx.x * blockDim.x + threadIdx.x;
  if (i >= (size_t)Nn * 96) return;
  const int n = (int)(i / 96), c = (int)(i % 96);
  float v = 0.0f;
  if (c < 64)      v = atom_emb[atom_idx[n] * 64 + c];
  else if (c < 80) v = peh[(size_t)n * 16 + (c - 64)];
  d[i] = f2bf_bits(v);
}
__global__ void pe_mlp_kernel(const float* __restrict__ pe, const float* __restrict__ w1,
                              const float* __restrict__ b1, const float* __restrict__ w2,
                              const float* __restrict__ b2, float* __restrict__ outp, int Nn) {
  const int n = blockIdx.x * blockDim.x + threadIdx.x;
  if (n >= Nn) return;
  float h1[16];
#pragma unroll
  for (int o = 0; o < 16; ++o) {
    float s = b1[o];
    for (int i = 0; i < 20; ++i) s += pe[(size_t)n * 20 + i] * w1[o * 20 + i];
    h1[o] = fmaxf(s, 0.0f);
  }
#pragma unroll
  for (int o = 0; o < 16; ++o) {
    float s = b2[o];
    for (int i = 0; i < 16; ++i) s += h1[i] * w2[o * 16 + i];
    outp[(size_t)n * 16 + o] = s;
  }
}
__global__ void build_inv_kernel(const int* __restrict__ batch, const int* __restrict__ pos,
                                 int* __restrict__ inv, int Nn) {
  const int i = blockIdx.x * blockDim.x + threadIdx.x;
  if (i < Nn) inv[batch[i] * LMAX + pos[i]] = i;
}
__global__ void edge_scatter_kernel(const float* __restrict__ x, const int* __restrict__ esrc,
                                    const int* __restrict__ edst, float* __restrict__ agg, int E) {
  size_t i = (size_t)blockIdx.x * blockDim.x + threadIdx.x;
  if (i >= (size_t)E * DMODEL) return;
  const int e = (int)(i >> 7), d = (int)(i & 127);
  atomicAdd(&agg[(size_t)edst[e] * DMODEL + d], x[(size_t)esrc[e] * DMODEL + d]);
}
__global__ void pool_kernel(const float* __restrict__ x, const int* __restrict__ batch,
                            float* __restrict__ pooled, int Nn) {
  size_t i = (size_t)blockIdx.x * blockDim.x + threadIdx.x;
  if (i >= (size_t)Nn * DMODEL) return;
  const int n = (int)(i >> 7), d = (int)(i & 127);
  atomicAdd(&pooled[(size_t)batch[n] * DMODEL + d], x[i]);
}
// training-mode BatchNorm1d (biased var): per-channel sums, then apply
__global__ __launch_bounds__(256) void bn_stats_kernel(const float* __restrict__ x,
                                                       float* __restrict__ stats, int M, int C) {
  const int c = blockIdx.x;
  float s = 0.f, s2 = 0.f;
  for (int r = threadIdx.x; r < M; r += 256) {
    const float v = x[(size_t)r * C + c];
    s += v; s2 += v * v;
  }
  __shared__ float ra[256], rb[256];
  ra[threadIdx.x] = s; rb[threadIdx.x] = s2;
  __syncthreads();
  for (int st = 128; st > 0; st >>= 1) {
    if (threadIdx.x < st) { ra[threadIdx.x] += ra[threadIdx.x + st]; rb[threadIdx.x] += rb[threadIdx.x + st]; }
    __syncthreads();
  }
  if (threadIdx.x == 0) { stats[c] = ra[0]; stats[C + c] = rb[0]; }
}
__global__ void bn_apply_kernel(float* __restrict__ x, const float* __restrict__ stats,
                                const float* __restrict__ g, const float* __restrict__ b,
                                int M, int C) {
  size_t i = (size_t)blockIdx.x * blockDim.x + threadIdx.x;
  if (i >= (size_t)M * C) return;
  const int c = (int)(i % C);
  const float mean = stats[c] / (float)M;
  const float var  = stats[C + c] / (float)M - mean * mean;
  x[i] = (x[i] - mean) * rsqrtf(var + 1e-5f) * g[c] + b[c];
}
__global__ __launch_bounds__(128) void readout_final_kernel(const float* __restrict__ hid,
                                                            const float* __restrict__ w2,
                                                            const float* __restrict__ b2,
                                                            float* __restrict__ out) {
  __shared__ float red[128];
  const int b = blockIdx.x, t = threadIdx.x;
  red[t] = hid[(size_t)b * DMODEL + t] * w2[t];
  __syncthreads();
  for (int s = 64; s > 0; s >>= 1) { if (t < s) red[t] += red[t + s]; __syncthreads(); }
  if (t == 0) out[b] = red[0] + b2[0];
}

// ---------------------------------------------------------------------------
// Host orchestration
// ---------------------------------------------------------------------------
static inline unsigned int cdiv(size_t a, size_t b) { return (unsigned int)((a + b - 1) / b); }

static void launch_gemm(const unsigned short* A, const unsigned short* W, const float* bias,
                        float* C, int M, int Nc, int K, int relu, hipStream_t s) {
  const int blks = ((M + 31) / 32) * (Nc / 32);
  wmma_gemm_bias_kernel<<<cdiv(blks, 4), 128, 0, s>>>(A, W, bias, C, M, Nc, K, relu);
}

extern "C" void kernel_launch(void* const* d_in, const int* in_sizes, int n_in,
                              void* d_out, int out_size, void* d_ws, size_t ws_size,
                              hipStream_t stream) {
  const int* atom_idx = (const int*)d_in[0];
  const float* pe     = (const float*)d_in[1];
  const int* edge_src = (const int*)d_in[2];
  const int* edge_dst = (const int*)d_in[3];
  const int* batch    = (const int*)d_in[4];
  const int* node_pos = (const int*)d_in[5];
  /* d_in[6] pad_mask unused (inv map encodes padding) */
  const float* atom_emb = (const float*)d_in[7];
  const float* pe_w1 = (const float*)d_in[8];
  const float* pe_b1 = (const float*)d_in[9];
  const float* pe_w2 = (const float*)d_in[10];
  const float* pe_b2 = (const float*)d_in[11];
  const float* pe_bn_g = (const float*)d_in[12];
  const float* pe_bn_b = (const float*)d_in[13];
  const float* in_w = (const float*)d_in[14];
  const float* in_b = (const float*)d_in[15];
  const float* gin_w1 = (const float*)d_in[16];
  const float* gin_b1 = (const float*)d_in[17];
  const float* gin_w2 = (const float*)d_in[18];
  const float* gin_b2 = (const float*)d_in[19];
  const float* gin_bn_g = (const float*)d_in[20];
  const float* gin_bn_b = (const float*)d_in[21];
  const float* qkv_w = (const float*)d_in[22];
  const float* qkv_b = (const float*)d_in[23];
  const float* out_w = (const float*)d_in[24];
  const float* out_b = (const float*)d_in[25];
  const float* attn_bn_g = (const float*)d_in[26];
  const float* attn_bn_b = (const float*)d_in[27];
  const float* ffn_w1 = (const float*)d_in[28];
  const float* ffn_b1 = (const float*)d_in[29];
  const float* ffn_w2 = (const float*)d_in[30];
  const float* ffn_b2 = (const float*)d_in[31];
  const float* ffn_bn_g = (const float*)d_in[32];
  const float* ffn_bn_b = (const float*)d_in[33];
  const float* ro_w1 = (const float*)d_in[34];
  const float* ro_b1 = (const float*)d_in[35];
  const float* ro_w2 = (const float*)d_in[36];
  const float* ro_b2 = (const float*)d_in[37];
  float* out = (float*)d_out;

  const int N = in_sizes[0];
  const int E = in_sizes[2];

  // ---- workspace carve-out -------------------------------------------------
  size_t off = 0;
  char* base = (char*)d_ws;
  auto carve = [&](size_t bytes) -> void* {
    void* p = base + off;
    off = (off + bytes + 255) & ~(size_t)255;
    return p;
  };
  float* x    = (float*)carve((size_t)N * DMODEL * 4);
  float* bufA = (float*)carve((size_t)N * 3 * DMODEL * 4);
  float* bufB = (float*)carve((size_t)N * DMODEL * 4);
  float* bufC = (float*)carve((size_t)N * DMODEL * 4);
  unsigned short* S = (unsigned short*)carve((size_t)N * FFDIM * 2);  // bf16 staging
  unsigned short* w_in  = (unsigned short*)carve((size_t)DMODEL * 96 * 2);
  unsigned short* w_g1  = (unsigned short*)carve((size_t)NLAYER * DMODEL * DMODEL * 2);
  unsigned short* w_g2  = (unsigned short*)carve((size_t)NLAYER * DMODEL * DMODEL * 2);
  unsigned short* w_qkv = (unsigned short*)carve((size_t)NLAYER * 3 * DMODEL * DMODEL * 2);
  unsigned short* w_out = (unsigned short*)carve((size_t)NLAYER * DMODEL * DMODEL * 2);
  unsigned short* w_f1  = (unsigned short*)carve((size_t)NLAYER * FFDIM * DMODEL * 2);
  unsigned short* w_f2  = (unsigned short*)carve((size_t)NLAYER * DMODEL * FFDIM * 2);
  unsigned short* w_ro1 = (unsigned short*)carve((size_t)DMODEL * DMODEL * 2);
  float* stats  = (float*)carve(2 * 3 * DMODEL * 4);
  int*   inv    = (int*)carve((size_t)BATCH * LMAX * 4);
  float* peh    = (float*)carve((size_t)N * 16 * 4);
  float* pooled = (float*)carve((size_t)BATCH * DMODEL * 4);
  float* rhid   = (float*)carve((size_t)BATCH * DMODEL * 4);
  (void)ws_size; (void)n_in; (void)out_size;

  // ---- one-time weight conversion to bf16 ---------------------------------
  struct Cv { const float* s; unsigned short* d; size_t n; };
  const Cv cvs[] = {
    { gin_w1, w_g1,  (size_t)NLAYER * DMODEL * DMODEL },
    { gin_w2, w_g2,  (size_t)NLAYER * DMODEL * DMODEL },
    { qkv_w,  w_qkv, (size_t)NLAYER * 3 * DMODEL * DMODEL },
    { out_w,  w_out, (size_t)NLAYER * DMODEL * DMODEL },
    { ffn_w1, w_f1,  (size_t)NLAYER * FFDIM * DMODEL },
    { ffn_w2, w_f2,  (size_t)NLAYER * DMODEL * FFDIM },
    { ro_w1,  w_ro1, (size_t)DMODEL * DMODEL },
  };
  for (const Cv& c : cvs)
    f32_to_bf16_kernel<<<cdiv(c.n, 256), 256, 0, stream>>>(c.s, c.d, c.n);
  pad_inw_kernel<<<cdiv(DMODEL * 96, 256), 256, 0, stream>>>(in_w, w_in);

  // ---- input encoder: PE MLP -> BN -> concat(pad K=96) -> linear ----------
  pe_mlp_kernel<<<cdiv(N, 256), 256, 0, stream>>>(pe, pe_w1, pe_b1, pe_w2, pe_b2, peh, N);
  bn_stats_kernel<<<16, 256, 0, stream>>>(peh, stats, N, 16);
  bn_apply_kernel<<<cdiv((size_t)N * 16, 256), 256, 0, stream>>>(peh, stats, pe_bn_g, pe_bn_b, N, 16);
  build_input_kernel<<<cdiv((size_t)N * 96, 256), 256, 0, stream>>>(atom_idx, atom_emb, peh, S, N);
  launch_gemm(S, w_in, in_b, x, N, DMODEL, 96, 0, stream);

  // slot -> node inverse map (padding = -1)
  fill_i32_kernel<<<cdiv((size_t)BATCH * LMAX, 256), 256, 0, stream>>>(inv, -1, (size_t)BATCH * LMAX);
  build_inv_kernel<<<cdiv(N, 256), 256, 0, stream>>>(batch, node_pos, inv, N);

  const size_t nD = (size_t)N * DMODEL;
  for (int l = 0; l < NLAYER; ++l) {
    // GIN: agg = segment_sum(x[src] -> dst); g = BN(mlp(x + agg))
    zero_f32_kernel<<<cdiv(nD, 256), 256, 0, stream>>>(bufC, nD);
    edge_scatter_kernel<<<cdiv((size_t)E * DMODEL, 256), 256, 0, stream>>>(x, edge_src, edge_dst, bufC, E);
    add_to_bf16_kernel<<<cdiv(nD, 256), 256, 0, stream>>>(x, bufC, S, nD);
    launch_gemm(S, w_g1 + (size_t)l * DMODEL * DMODEL, gin_b1 + l * DMODEL, bufA, N, DMODEL, DMODEL, 1, stream);
    f32_to_bf16_kernel<<<cdiv(nD, 256), 256, 0, stream>>>(bufA, S, nD);
    launch_gemm(S, w_g2 + (size_t)l * DMODEL * DMODEL, gin_b2 + l * DMODEL, bufB, N, DMODEL, DMODEL, 0, stream);
    bn_stats_kernel<<<DMODEL, 256, 0, stream>>>(bufB, stats, N, DMODEL);
    bn_apply_kernel<<<cdiv(nD, 256), 256, 0, stream>>>(bufB, stats, gin_bn_g + l * DMODEL, gin_bn_b + l * DMODEL, N, DMODEL);

    // attention: qkv -> per-graph masked MHA -> out-proj -> BN
    f32_to_bf16_kernel<<<cdiv(nD, 256), 256, 0, stream>>>(x, S, nD);
    launch_gemm(S, w_qkv + (size_t)l * 3 * DMODEL * DMODEL, qkv_b + l * 3 * DMODEL, bufA, N, 3 * DMODEL, DMODEL, 0, stream);
    attention_kernel<<<BATCH, 128, 0, stream>>>(bufA, inv, bufC);
    f32_to_bf16_kernel<<<cdiv(nD, 256), 256, 0, stream>>>(bufC, S, nD);
    launch_gemm(S, w_out + (size_t)l * DMODEL * DMODEL, out_b + l * DMODEL, bufC, N, DMODEL, DMODEL, 0, stream);
    bn_stats_kernel<<<DMODEL, 256, 0, stream>>>(bufC, stats, N, DMODEL);
    bn_apply_kernel<<<cdiv(nD, 256), 256, 0, stream>>>(bufC, stats, attn_bn_g + l * DMODEL, attn_bn_b + l * DMODEL, N, DMODEL);

    // x = x + relu(g) + o
    residual3_kernel<<<cdiv(nD, 256), 256, 0, stream>>>(x, bufB, bufC, nD);

    // FFN + BN residual
    f32_to_bf16_kernel<<<cdiv(nD, 256), 256, 0, stream>>>(x, S, nD);
    launch_gemm(S, w_f1 + (size_t)l * FFDIM * DMODEL, ffn_b1 + l * FFDIM, bufA, N, FFDIM, DMODEL, 1, stream);
    f32_to_bf16_kernel<<<cdiv((size_t)N * FFDIM, 256), 256, 0, stream>>>(bufA, S, (size_t)N * FFDIM);
    launch_gemm(S, w_f2 + (size_t)l * DMODEL * FFDIM, ffn_b2 + l * DMODEL, bufB, N, DMODEL, FFDIM, 0, stream);
    bn_stats_kernel<<<DMODEL, 256, 0, stream>>>(bufB, stats, N, DMODEL);
    bn_apply_kernel<<<cdiv(nD, 256), 256, 0, stream>>>(bufB, stats, ffn_bn_g + l * DMODEL, ffn_bn_b + l * DMODEL, N, DMODEL);
    add_inplace_kernel<<<cdiv(nD, 256), 256, 0, stream>>>(x, bufB, nD);
  }

  // ---- global_add_pool + readout MLP --------------------------------------
  zero_f32_kernel<<<cdiv((size_t)BATCH * DMODEL, 256), 256, 0, stream>>>(pooled, (size_t)BATCH * DMODEL);
  pool_kernel<<<cdiv(nD, 256), 256, 0, stream>>>(x, batch, pooled, N);
  f32_to_bf16_kernel<<<cdiv((size_t)BATCH * DMODEL, 256), 256, 0, stream>>>(pooled, S, (size_t)BATCH * DMODEL);
  launch_gemm(S, w_ro1, ro_b1, rhid, BATCH, DMODEL, DMODEL, 1, stream);
  readout_final_kernel<<<BATCH, 128, 0, stream>>>(rhid, ro_w2, ro_b2, out);
}
